// VSSSBlock1D_17592186044631
// MI455X (gfx1250) — matrix operand order repaired and checked by reading, not compiled, not run on hardware
//
#include <hip/hip_runtime.h>
#include <hip/hip_bf16.h>
#include <math.h>

// ---------------------------------------------------------------------------
// Mamba / S6 block for MI455X (gfx1250, wave32, WMMA).
// GEMMs: v_wmma_f32_16x16x32_bf16, hi/lo bf16 split compensation (3 WMMAs),
// fragment-ordered LDS so fragment reads are 2x ds_load_b128 per lane.
// Scan: 8-way chunked (3-pass) diagonal scan for occupancy; z-gating fused.
// ---------------------------------------------------------------------------

typedef __attribute__((ext_vector_type(16))) __bf16 v16bf;
typedef __attribute__((ext_vector_type(8)))  float  v8f;

#define LSEQ 4096
#define BM 64
#define BN 128
#define BK 32
#define NCH 8
#define CHL (LSEQ / NCH)   // 512

__device__ __forceinline__ unsigned pack2(float a, float b) {
  union { __bf16 h[2]; unsigned u; } un;
  un.h[0] = (__bf16)a; un.h[1] = (__bf16)b;
  return un.u;
}
__device__ __forceinline__ float lofrom(float v) {
  return v - (float)((__bf16)v);
}

// out[b,o,l] = sum_k W[o,k] * act[b,k,l]  (+ epilogue by mode)
// mode 0: +bias, split rows: o<1024 -> out0 (u_raw), else out1 (z)
// mode 1: plain write
// mode 2: softplus(acc + 2*bias)  (dt_proj: reference adds bias twice)
// mode 3: +bias + resid[b,o,l]    (out_proj + residual)
__global__ __launch_bounds__(256) void gemm_wmma_split(
    const float* __restrict__ W, const float* __restrict__ act,
    float* __restrict__ out0, float* __restrict__ out1,
    const float* __restrict__ bias, const float* __restrict__ resid,
    int O, int K, long actBatchStride, long outBatchStride,
    long residBatchStride, int mode)
{
  // Fragment-ordered tiles: [tile][lane][elem-pair(u32)]
  __shared__ __align__(16) unsigned sAhi[4][32][8];   // 4 row-tiles of 16
  __shared__ __align__(16) unsigned sAlo[4][32][8];
  __shared__ __align__(16) unsigned sBhi[8][32][8];   // 8 col-tiles of 16
  __shared__ __align__(16) unsigned sBlo[8][32][8];

  const int tid = threadIdx.x;
  const int lane = tid & 31;
  const int wave = tid >> 5;
  const int warpRow = wave >> 2;   // 0..1 -> 32 rows
  const int warpCol = wave & 3;    // 0..3 -> 32 cols
  const int b = blockIdx.z;
  const int oBlock = blockIdx.y * BM;
  const int lBlock = blockIdx.x * BN;

  const float* actB = act + (long)b * actBatchStride;

  v8f acc[2][2];
  #pragma unroll
  for (int r = 0; r < 2; ++r)
    #pragma unroll
    for (int c = 0; c < 2; ++c)
      #pragma unroll
      for (int j = 0; j < 8; ++j) acc[r][c][j] = 0.f;

  for (int k0 = 0; k0 < K; k0 += BK) {
    // ---- stage W tile (BM x BK): float4 over k, scatter to fragment order
    // A-layout inverse: tile=r>>4, lane=((c>>3)&1)*16 + (r&15),
    //                   e=(c&7)|((c>>4)<<3)
    #pragma unroll
    for (int i = 0; i < 2; ++i) {          // 512 quads total
      int q = tid + i * 256;
      int r = q >> 3;                      // 0..63
      int c = (q & 7) * 4;                 // 0,4,...,28
      float4 v = *reinterpret_cast<const float4*>(
          &W[(long)(oBlock + r) * K + (k0 + c)]);
      int t = r >> 4;
      int ln = ((c >> 3) & 1) * 16 + (r & 15);
      int ep = (((c & 7) | ((c >> 4) << 3)) >> 1);   // even
      uint2 hi = make_uint2(pack2(v.x, v.y), pack2(v.z, v.w));
      uint2 lo = make_uint2(pack2(lofrom(v.x), lofrom(v.y)),
                            pack2(lofrom(v.z), lofrom(v.w)));
      *reinterpret_cast<uint2*>(&sAhi[t][ln][ep]) = hi;
      *reinterpret_cast<uint2*>(&sAlo[t][ln][ep]) = lo;
    }
    // ---- stage act tile (BK x BN): octets = 4 k-rows x 2 n-cols
    // B-layout inverse: tile=n>>4, lane=((k>>3)&1)*16 + (n&15), e=(k&7)|((k>>4)<<3)
    #pragma unroll
    for (int i = 0; i < 2; ++i) {          // 512 octets total
      int q = tid + i * 256;
      int kg = q >> 6;                     // 0..7 -> k = kg*4
      int n2 = (q & 63) * 2;               // 0..126
      int k = kg * 4;
      float2 r0 = *reinterpret_cast<const float2*>(&actB[(long)(k0 + k + 0) * LSEQ + lBlock + n2]);
      float2 r1 = *reinterpret_cast<const float2*>(&actB[(long)(k0 + k + 1) * LSEQ + lBlock + n2]);
      float2 r2 = *reinterpret_cast<const float2*>(&actB[(long)(k0 + k + 2) * LSEQ + lBlock + n2]);
      float2 r3 = *reinterpret_cast<const float2*>(&actB[(long)(k0 + k + 3) * LSEQ + lBlock + n2]);
      int ep = (((k & 7) | ((k >> 4) << 3)) >> 1);   // 0,2,4,6
      #pragma unroll
      for (int j = 0; j < 2; ++j) {
        int n = n2 + j;
        int t = n >> 4;
        int ln = ((k >> 3) & 1) * 16 + (n & 15);
        float a0 = j ? r0.y : r0.x, a1 = j ? r1.y : r1.x;
        float a2 = j ? r2.y : r2.x, a3 = j ? r3.y : r3.x;
        *reinterpret_cast<uint2*>(&sBhi[t][ln][ep]) =
            make_uint2(pack2(a0, a1), pack2(a2, a3));
        *reinterpret_cast<uint2*>(&sBlo[t][ln][ep]) =
            make_uint2(pack2(lofrom(a0), lofrom(a1)),
                       pack2(lofrom(a2), lofrom(a3)));
      }
    }
    __syncthreads();

    // ---- fragment reads: contiguous 32B per lane (2x ds_load_b128)
    v16bf ahi[2], alo[2], bhi[2], blo[2];
    #pragma unroll
    for (int r = 0; r < 2; ++r) {
      int t = warpRow * 2 + r;
      ahi[r] = *reinterpret_cast<const v16bf*>(&sAhi[t][lane][0]);
      alo[r] = *reinterpret_cast<const v16bf*>(&sAlo[t][lane][0]);
    }
    #pragma unroll
    for (int c = 0; c < 2; ++c) {
      int t = warpCol * 2 + c;
      bhi[c] = *reinterpret_cast<const v16bf*>(&sBhi[t][lane][0]);
      blo[c] = *reinterpret_cast<const v16bf*>(&sBlo[t][lane][0]);
    }

    // hi*hi + lo*hi + hi*lo : fp32-comparable accuracy from bf16 WMMA
    #pragma unroll
    for (int r = 0; r < 2; ++r)
      #pragma unroll
      for (int c = 0; c < 2; ++c) {
        acc[r][c] = __builtin_amdgcn_wmma_f32_16x16x32_bf16(
            false, ahi[r], false, bhi[c], (short)0, acc[r][c], false, false);
        acc[r][c] = __builtin_amdgcn_wmma_f32_16x16x32_bf16(
            false, alo[r], false, bhi[c], (short)0, acc[r][c], false, false);
        acc[r][c] = __builtin_amdgcn_wmma_f32_16x16x32_bf16(
            false, ahi[r], false, blo[c], (short)0, acc[r][c], false, false);
      }
    __syncthreads();
  }

  // Epilogue. D layout: n = lane&15 ; vgpr j -> m = j + (lane>=16 ? 8 : 0)
  const int nD = lane & 15;
  const int mD = (lane >> 4) * 8;
  #pragma unroll
  for (int r = 0; r < 2; ++r)
    #pragma unroll
    for (int c = 0; c < 2; ++c)
      #pragma unroll
      for (int j = 0; j < 8; ++j) {
        int o = oBlock + warpRow * 32 + r * 16 + mD + j;
        int l = lBlock + warpCol * 32 + c * 16 + nD;
        float v = acc[r][c][j];
        long outIdx = (long)b * outBatchStride + (long)o * LSEQ + l;
        if (mode == 0) {
          v += bias[o];
          if (o < 1024) out0[(long)b * outBatchStride + (long)o * LSEQ + l] = v;
          else          out1[(long)b * outBatchStride + (long)(o - 1024) * LSEQ + l] = v;
        } else if (mode == 1) {
          out0[outIdx] = v;
        } else if (mode == 2) {
          float s = v + 2.f * bias[o];      // reference adds dt_proj_b twice
          float sp = (s > 0.f) ? (s + log1pf(__expf(-s))) : log1pf(__expf(s));
          out0[outIdx] = sp;
        } else {
          v += bias[o] + resid[(long)b * residBatchStride + (long)o * LSEQ + l];
          out0[outIdx] = v;
        }
      }
}

// depthwise conv1d (k=3, pad=1) + SiLU
__global__ __launch_bounds__(256) void conv_silu_kernel(
    const float* __restrict__ uin, const float* __restrict__ cw,
    const float* __restrict__ cb, float* __restrict__ uout)
{
  long i = (long)blockIdx.x * 256 + threadIdx.x;   // over 8*1024*4096
  int l = (int)(i & (LSEQ - 1));
  int d = (int)((i >> 12) & 1023);
  float xm = (l > 0)        ? uin[i - 1] : 0.f;
  float x0 = uin[i];
  float xp = (l < LSEQ - 1) ? uin[i + 1] : 0.f;
  float s = cw[d * 3 + 0] * xm + cw[d * 3 + 1] * x0 + cw[d * 3 + 2] * xp + cb[d];
  uout[i] = s / (1.f + __expf(-s));
}

// ---- chunked selective scan (3 passes) --------------------------------
// Pass 1: per (b,d,chunk): S = h starting from 0 over the chunk, P = prod dA.
__global__ __launch_bounds__(256) void scan_chunk_kernel(
    const float* __restrict__ delta, const float* __restrict__ u,
    const float* __restrict__ xdbl,  const float* __restrict__ A_log,
    float* __restrict__ P, float* __restrict__ S)
{
  __shared__ float sB[16][128];
  const int tid = threadIdx.x;
  const int d = blockIdx.x * 256 + tid;
  const int ch = blockIdx.y;
  const int b = blockIdx.z;
  const long gt = (long)b * 1024 + d;

  const float* dl = delta + gt * LSEQ;
  const float* uu = u + gt * LSEQ;
  const float* Bp = xdbl + ((long)b * 64 + 32) * LSEQ;

  float a2[16], h[16], p[16];
  #pragma unroll
  for (int n = 0; n < 16; ++n) {
    a2[n] = -__expf(A_log[d * 16 + n]) * 1.44269504f;
    h[n] = 0.f; p[n] = 1.f;
  }

  for (int l0 = ch * CHL; l0 < (ch + 1) * CHL; l0 += 128) {
    __syncthreads();
    #pragma unroll
    for (int i = 0; i < 8; ++i) {
      int idx = tid + i * 256;
      sB[idx >> 7][idx & 127] = Bp[(long)(idx >> 7) * LSEQ + l0 + (idx & 127)];
    }
    __syncthreads();
    for (int ll = 0; ll < 128; ++ll) {
      float dlt = dl[l0 + ll];
      float du = dlt * uu[l0 + ll];
      #pragma unroll
      for (int n = 0; n < 16; ++n) {
        float dA = exp2f(dlt * a2[n]);
        h[n] = dA * h[n] + du * sB[n][ll];
        p[n] *= dA;
      }
    }
  }
  long o = ((gt * NCH + ch) << 4);
  #pragma unroll
  for (int n = 0; n < 16; ++n) { P[o + n] = p[n]; S[o + n] = h[n]; }
}

// Pass 2: sequential combine over chunks -> initial state per chunk.
__global__ __launch_bounds__(256) void scan_combine_kernel(
    const float* __restrict__ P, const float* __restrict__ S,
    float* __restrict__ Hin)
{
  long idx = (long)blockIdx.x * 256 + threadIdx.x;  // gt*16 + n, 131072 total
  long gt = idx >> 4;
  int n = idx & 15;
  float h = 0.f;
  #pragma unroll
  for (int ch = 0; ch < NCH; ++ch) {
    long o = ((gt * NCH + ch) << 4) + n;
    Hin[o] = h;
    h = P[o] * h + S[o];
  }
}

// Pass 3: re-run chunk from its true initial state; fuse D*u and z-gating.
__global__ __launch_bounds__(256) void scan_final_kernel(
    const float* __restrict__ delta, const float* __restrict__ u,
    const float* __restrict__ xdbl,  const float* __restrict__ z,
    const float* __restrict__ A_log, const float* __restrict__ Dp,
    const float* __restrict__ Hin,   float* __restrict__ y)
{
  __shared__ float sB[16][128];
  __shared__ float sC[16][128];
  const int tid = threadIdx.x;
  const int d = blockIdx.x * 256 + tid;
  const int ch = blockIdx.y;
  const int b = blockIdx.z;
  const long gt = (long)b * 1024 + d;

  const float* dl = delta + gt * LSEQ;
  const float* uu = u + gt * LSEQ;
  const float* zz = z + gt * LSEQ;
  float*       yy = y + gt * LSEQ;
  const float* Bp = xdbl + ((long)b * 64 + 32) * LSEQ;
  const float* Cp = xdbl + ((long)b * 64 + 48) * LSEQ;

  float a2[16], h[16];
  long ho = ((gt * NCH + ch) << 4);
  #pragma unroll
  for (int n = 0; n < 16; ++n) {
    a2[n] = -__expf(A_log[d * 16 + n]) * 1.44269504f;
    h[n] = Hin[ho + n];
  }
  const float Dd = Dp[d];

  for (int l0 = ch * CHL; l0 < (ch + 1) * CHL; l0 += 128) {
    __syncthreads();
    #pragma unroll
    for (int i = 0; i < 8; ++i) {
      int idx = tid + i * 256;
      int n = idx >> 7, ll = idx & 127;
      sB[n][ll] = Bp[(long)n * LSEQ + l0 + ll];
      sC[n][ll] = Cp[(long)n * LSEQ + l0 + ll];
    }
    __syncthreads();
    for (int ll = 0; ll < 128; ++ll) {
      int l = l0 + ll;
      float dlt = dl[l];
      float ul = uu[l];
      float du = dlt * ul;
      float yacc = 0.f;
      #pragma unroll
      for (int n = 0; n < 16; ++n) {
        float dA = exp2f(dlt * a2[n]);
        h[n] = dA * h[n] + du * sB[n][ll];
        yacc += h[n] * sC[n][ll];
      }
      float zl = zz[l];
      float sz = zl / (1.f + __expf(-zl));    // silu(z)
      yy[l] = (yacc + Dd * ul) * sz;
    }
  }
}

extern "C" void kernel_launch(void* const* d_in, const int* in_sizes, int n_in,
                              void* d_out, int out_size, void* d_ws, size_t ws_size,
                              hipStream_t stream) {
  const float* x         = (const float*)d_in[0];   // (8,512,4096)
  const float* in_proj_w = (const float*)d_in[1];   // (2048,512)
  const float* in_proj_b = (const float*)d_in[2];   // (2048)
  const float* conv_w    = (const float*)d_in[3];   // (1024,1,3)
  const float* conv_b    = (const float*)d_in[4];   // (1024)
  const float* x_proj_w  = (const float*)d_in[5];   // (64,1024)
  const float* dt_proj_w = (const float*)d_in[6];   // (1024,32)
  const float* dt_proj_b = (const float*)d_in[7];   // (1024)
  const float* A_log     = (const float*)d_in[8];   // (1024,16)
  const float* Dvec      = (const float*)d_in[9];   // (1024)
  const float* out_proj_w= (const float*)d_in[10];  // (512,1024)
  const float* out_proj_b= (const float*)d_in[11];  // (512)
  float* out = (float*)d_out;                       // (8,512,4096)

  const long N = 8L * 1024 * LSEQ;                  // 33.5M elems
  const long M = 8L * 1024 * NCH * 16;              // 1M elems
  float* u_raw = (float*)d_ws;                      // also reused for y
  float* zbuf  = u_raw + N;
  float* ubuf  = zbuf + N;
  float* delta = ubuf + N;
  float* xdbl  = delta + N;                         // (8,64,4096)
  float* Pbuf  = xdbl + 8L * 64 * LSEQ;
  float* Sbuf  = Pbuf + M;
  float* Hin   = Sbuf + M;

  dim3 blk(256);

  // 1) in_proj: xz = W(2048x512) @ x ; split -> u_raw, z
  gemm_wmma_split<<<dim3(LSEQ / BN, 2048 / BM, 8), blk, 0, stream>>>(
      in_proj_w, x, u_raw, zbuf, in_proj_b, nullptr,
      2048, 512, 512L * LSEQ, 1024L * LSEQ, 0, 0);

  // 2) depthwise conv + SiLU: u_raw -> u
  conv_silu_kernel<<<(unsigned)(N / 256), blk, 0, stream>>>(u_raw, conv_w, conv_b, ubuf);

  // 3) x_proj: x_dbl = W(64x1024) @ u
  gemm_wmma_split<<<dim3(LSEQ / BN, 64 / BM, 8), blk, 0, stream>>>(
      x_proj_w, ubuf, xdbl, nullptr, nullptr, nullptr,
      64, 1024, 1024L * LSEQ, 64L * LSEQ, 0, 1);

  // 4) dt_proj: delta = softplus(W(1024x32) @ x_dbl[:32] + 2*dt_proj_b)
  gemm_wmma_split<<<dim3(LSEQ / BN, 1024 / BM, 8), blk, 0, stream>>>(
      dt_proj_w, xdbl, delta, nullptr, dt_proj_b, nullptr,
      1024, 32, 64L * LSEQ, 1024L * LSEQ, 0, 2);

  // 5) chunked selective scan (+ D*u skip + z gating), y -> u_raw (reuse)
  scan_chunk_kernel<<<dim3(4, NCH, 8), blk, 0, stream>>>(
      delta, ubuf, xdbl, A_log, Pbuf, Sbuf);
  scan_combine_kernel<<<512, blk, 0, stream>>>(Pbuf, Sbuf, Hin);
  scan_final_kernel<<<dim3(4, NCH, 8), blk, 0, stream>>>(
      delta, ubuf, xdbl, zbuf, A_log, Dvec, Hin, u_raw);

  // 6) out_proj: out = W(512x1024) @ y + bias + residual(x)
  gemm_wmma_split<<<dim3(LSEQ / BN, 512 / BM, 8), blk, 0, stream>>>(
      out_proj_w, u_raw, out, nullptr, out_proj_b, x,
      512, 1024, 1024L * LSEQ, 512L * LSEQ, 512L * LSEQ, 3);
}
